// Model_9552007266582
// MI455X (gfx1250) — matrix operand order repaired and checked by reading, not compiled, not run on hardware
//
#include <hip/hip_runtime.h>
#include <hip/hip_bf16.h>

// ---------------------------------------------------------------------------
// Model constants
// ---------------------------------------------------------------------------
#define BATCH     4
#define SEQ_LEN   1024
#define LSEQ      1025            // SEQ_LEN + 1
#define ENC_IN    7
#define MARK      4
#define D_MODEL   512
#define N_HEADS   8
#define D_K       64
#define D_FF      2048
#define E_LAYERS  4
#define D_BN      128
#define NTOT      1361            // 1025 + 256 + 64 + 16
#define NTOT_PAD  1376            // padded leading dim for transposed V (16B aligned)
#define PRED_OUT  672             // 96 * 7

typedef _Float16 half_t;
typedef __attribute__((ext_vector_type(16))) _Float16 v16h;
typedef __attribute__((ext_vector_type(8)))  _Float16 v8h;
typedef __attribute__((ext_vector_type(8)))  float    v8f;

#define SHUF16(a, b) __builtin_shufflevector(a, b, 0,1,2,3,4,5,6,7,8,9,10,11,12,13,14,15)

// ---------------------------------------------------------------------------
// f32 -> f16 casts
// ---------------------------------------------------------------------------
__global__ void cast_f32_f16(const float* __restrict__ in, half_t* __restrict__ out, long long n) {
    long long i = (long long)blockIdx.x * 256 + threadIdx.x;
    if (i < n) out[i] = (half_t)in[i];
}

// rows with per-batch stride: row r -> in + (r/rpb)*sb + (r%rpb)*cols
__global__ void cast_f32_f16_strided(const float* __restrict__ in, half_t* __restrict__ out,
                                     int rows, int cols, int rpb, long long sb) {
    long long i = (long long)blockIdx.x * 256 + threadIdx.x;
    long long n = (long long)rows * cols;
    if (i >= n) return;
    int c = (int)(i % cols);
    int r = (int)(i / cols);
    long long src = (long long)(r / rpb) * sb + (long long)(r % rpb) * cols + c;
    out[i] = (half_t)in[src];
}

// ---------------------------------------------------------------------------
// Pack a K x N f32 weight into WMMA B-fragment order, f16, N padded to Npad
// (multiple of 64, zero-filled). Layout:
//   out[(((kt*(Npad/16) + ntile)*32 + lane)*16 + e]
//   holds W[kt*32 + (lane&16 ? 16 : 0) + e][ntile*16 + (lane&15)]
// so a GEMM wave reads each fragment as two contiguous 16B chunks per lane.
// ---------------------------------------------------------------------------
__global__ void pack_b_kernel(const float* __restrict__ in, half_t* __restrict__ out,
                              int K, int N, int Npad) {
    long long i = (long long)blockIdx.x * 256 + threadIdx.x;
    long long total = (long long)K * Npad;
    if (i >= total) return;
    int e = (int)(i & 15);
    long long t = i >> 4;
    int lane = (int)(t & 31); t >>= 5;
    int nt16 = Npad >> 4;
    int ntile = (int)(t % nt16);
    int kt = (int)(t / nt16);
    int kk  = kt * 32 + ((lane & 16) ? 16 : 0) + e;
    int col = ntile * 16 + (lane & 15);
    float v = (col < N && kk < K) ? in[(long long)kk * N + col] : 0.0f;
    out[i] = (half_t)v;
}

// ---------------------------------------------------------------------------
// Pyraformer mask (blocked = !allow), N=1361, windows (4,4,4), inner=5
// ---------------------------------------------------------------------------
__global__ void build_mask_kernel(unsigned char* __restrict__ blocked) {
    long long idx = (long long)blockIdx.x * 256 + threadIdx.x;
    const long long total = (long long)NTOT * NTOT;
    if (idx >= total) return;
    int i = (int)(idx / NTOT), j = (int)(idx % NTOT);
    const int st[5] = {0, 1025, 1281, 1345, 1361};
    int li = (i < 1025) ? 0 : (i < 1281) ? 1 : (i < 1345) ? 2 : 3;
    int lj = (j < 1025) ? 0 : (j < 1281) ? 1 : (j < 1345) ? 2 : 3;
    bool allow = false;
    int d = (i > j) ? (i - j) : (j - i);
    if (li == lj && d <= 2) allow = true;
    if (li == lj + 1) {                       // i parent, j child
        int p  = i - st[li];
        int lo = st[lj] + p * 4;
        int hi = (i == st[li + 1] - 1) ? st[li] : st[lj] + (p + 1) * 4;
        if (j >= lo && j < hi) allow = true;
    }
    if (lj == li + 1) {                       // j parent, i child
        int p  = j - st[lj];
        int lo = st[li] + p * 4;
        int hi = (j == st[lj + 1] - 1) ? st[lj] : st[li] + (p + 1) * 4;
        if (i >= lo && i < hi) allow = true;
    }
    blocked[idx] = allow ? 0 : 1;
}

// ---------------------------------------------------------------------------
// Embedding: token conv(3, circular-ish pad) + temporal + sinusoidal PE
// ---------------------------------------------------------------------------
__global__ void embed_kernel(const float* __restrict__ x_enc,
                             const float* __restrict__ xme,
                             const float* __restrict__ xmd,
                             const float* __restrict__ tw,  const float* __restrict__ tb,
                             const float* __restrict__ tmw, const float* __restrict__ tmb,
                             float* __restrict__ seq) {
    long long idx = (long long)blockIdx.x * 256 + threadIdx.x;
    const long long total = (long long)BATCH * LSEQ * D_MODEL;
    if (idx >= total) return;
    int d = (int)(idx % D_MODEL);
    long long bt = idx / D_MODEL;
    int t = (int)(bt % LSEQ);
    int b = (int)(bt / LSEQ);

    float acc = tb[d];
    #pragma unroll
    for (int k = 0; k < 3; ++k) {
        int p = t + k;  // padded index 0..1026
        #pragma unroll
        for (int c = 0; c < ENC_IN; ++c) {
            float xv = 0.0f;
            if (p >= 1 && p <= SEQ_LEN) xv = x_enc[((long long)b * SEQ_LEN + (p - 1)) * ENC_IN + c];
            else if (p == LSEQ + 1)     xv = x_enc[((long long)b * SEQ_LEN) * ENC_IN + c];
            acc += tw[(d * ENC_IN + c) * 3 + k] * xv;
        }
    }
    #pragma unroll
    for (int m = 0; m < MARK; ++m) {
        float xm = (t < SEQ_LEN) ? xme[((long long)b * SEQ_LEN + t) * MARK + m]
                                 : xmd[((long long)b * 96) * MARK + m];
        acc += xm * tmw[m * D_MODEL + d];
    }
    acc += tmb[d];
    int i2 = (d >> 1) << 1;
    float div = __expf(-logf(10000.0f) * (float)i2 / (float)D_MODEL);
    float ang = (float)t * div;
    acc += (d & 1) ? cosf(ang) : sinf(ang);
    seq[((long long)b * NTOT + t) * D_MODEL + d] = acc;
}

// ---------------------------------------------------------------------------
// Strided conv (kernel 4, stride 4) + ELU; layouts (b, t, channel)
// ---------------------------------------------------------------------------
__global__ void conv_elu_kernel(const float* __restrict__ in, const float* __restrict__ w,
                                const float* __restrict__ bias, float* __restrict__ out,
                                int outLen, long long inBStride, long long outBStride, long long outOff) {
    long long idx = (long long)blockIdx.x * 256 + threadIdx.x;
    long long total = (long long)BATCH * outLen * D_BN;
    if (idx >= total) return;
    int co = (int)(idx % D_BN);
    long long bt = idx / D_BN;
    int t = (int)(bt % outLen);
    int b = (int)(bt / outLen);
    float acc = bias[co];
    const float* inb = in + (long long)b * inBStride;
    #pragma unroll
    for (int k = 0; k < 4; ++k) {
        const float* row = inb + (long long)(4 * t + k) * D_BN;
        for (int ci = 0; ci < D_BN; ++ci)
            acc += w[(co * D_BN + ci) * 4 + k] * row[ci];
    }
    acc = (acc > 0.0f) ? acc : (__expf(acc) - 1.0f);
    out[(long long)b * outBStride + outOff + (long long)t * D_BN + co] = acc;
}

// ---------------------------------------------------------------------------
// WMMA GEMM: C[M,N] = A[M,K](f16, row-major) * Bp(K x Npad, pre-packed)
// Branchless K-loop: out-of-range rows clamped (stores guarded). Per wave:
// 16x64 tile via v_wmma_f32_16x16x32_f16; all operand loads are b128.
// Optional: bias, exact GELU, f32 out, f16 out, f16 transposed out (ldT).
// ---------------------------------------------------------------------------
__global__ __launch_bounds__(128)
void gemm_f16_wmma(const half_t* __restrict__ A, const half_t* __restrict__ Bp,
                   const float* __restrict__ bias,
                   float* __restrict__ Cf, half_t* __restrict__ Ch,
                   half_t* __restrict__ ChT, long long ldT, long long sbCT,
                   int M, int N, int K,
                   int rpbA, long long sbA, int rpbC, long long sbC, int act) {
    const int nt16 = ((N + 63) & ~63) >> 4;       // padded N in 16-wide tiles
    int wave = threadIdx.x >> 5;
    int lane = threadIdx.x & 31;
    int l16  = lane & 15;
    int hi   = (lane >> 4) & 1;
    int n0   = blockIdx.x * 64;
    int m0   = blockIdx.y * 64 + wave * 16;

    int rA = m0 + l16; if (rA > M - 1) rA = M - 1;            // clamp: always valid
    const half_t* aRow = A + (long long)(rA / rpbA) * sbA + (long long)(rA % rpbA) * K;
    const half_t* bBase = Bp + (((long long)(n0 >> 4) * 32) + lane) * 16;
    const long long bKStride = (long long)nt16 * 32 * 16;     // advance one kt

    v8f acc[4] = {};
    for (int k0 = 0; k0 < K; k0 += 32) {
        const v8h* ap = (const v8h*)(aRow + k0 + hi * 16);
        v16h a = SHUF16(ap[0], ap[1]);
        const half_t* bk = bBase + (long long)(k0 >> 5) * bKStride;
        #pragma unroll
        for (int nt = 0; nt < 4; ++nt) {
            const v8h* bp = (const v8h*)(bk + (long long)nt * 32 * 16);
            v16h bf = SHUF16(bp[0], bp[1]);
            acc[nt] = __builtin_amdgcn_wmma_f32_16x16x32_f16(
                false, a, false, bf, (short)0, acc[nt], false, false);
        }
    }

    #pragma unroll
    for (int nt = 0; nt < 4; ++nt) {
        int col = n0 + nt * 16 + l16;
        if (col >= N) continue;
        float bv = bias ? bias[col] : 0.0f;
        #pragma unroll
        for (int g = 0; g < 8; ++g) {
            int r = m0 + g + hi * 8;
            if (r < M) {
                float v = acc[nt][g] + bv;
                if (act == 1) v = 0.5f * v * (1.0f + erff(v * 0.70710678f));   // exact GELU
                long long rb = (long long)(r / rpbC);
                long long rr = (long long)(r % rpbC);
                if (Cf)  Cf [rb * sbC  + rr * N + col] = v;
                if (Ch)  Ch [rb * sbC  + rr * N + col] = (half_t)v;
                if (ChT) ChT[rb * sbCT + (long long)col * ldT + rr] = (half_t)v;
            }
        }
    }
}

// ---------------------------------------------------------------------------
// Flash attention, wave32, WMMA for QK^T and PV. One wave per (b, h, 16-query
// tile). Branchless operand loads (clamped addresses, b128 chunks); V is read
// from a transposed (d-major, ld=1376) buffer so PV B-fragments are contiguous.
// Masked scores = -1e9 exactly like the reference; streaming rescale
// exp(-1e9 - m) underflows to 0 so fully-masked prefixes contribute nothing.
// ---------------------------------------------------------------------------
__global__ __launch_bounds__(32)
void attn_wmma(const half_t* __restrict__ qH, const half_t* __restrict__ kH,
               const half_t* __restrict__ vT, const unsigned char* __restrict__ blocked,
               half_t* __restrict__ oH) {
    __shared__ half_t Plds[16 * 32];
    int lane = threadIdx.x & 31;
    int l16  = lane & 15;
    int hi   = (lane >> 4) & 1;
    int q0   = blockIdx.x * 16;
    int b    = blockIdx.y >> 3;
    int h    = blockIdx.y & 7;
    long long base = (long long)b * NTOT * D_MODEL + h * D_K;

    // preload Q fragments (16 x 64 -> 2 A-frags of K=32), clamped row
    int qrow = q0 + l16; int qrc = (qrow < NTOT) ? qrow : (NTOT - 1);
    const half_t* qp = qH + base + (long long)qrc * D_MODEL;
    v16h qa[2];
    #pragma unroll
    for (int kc = 0; kc < 2; ++kc) {
        const v8h* ap = (const v8h*)(qp + kc * 32 + hi * 16);
        qa[kc] = SHUF16(ap[0], ap[1]);
    }

    float m_[8], l_[8];
    v8f O[4] = {};
    #pragma unroll
    for (int g = 0; g < 8; ++g) { m_[g] = -1e30f; l_[g] = 0.0f; }

    const float scale = 0.125f;                   // 1/sqrt(64)
    const int nkb = (NTOT + 31) / 32;
    for (int kb = 0; kb < nkb; ++kb) {
        // S = Q * K^T for 32 keys -> two 16x16 f32 fragments
        v8f S[2] = {};
        #pragma unroll
        for (int t = 0; t < 2; ++t) {
            int key = kb * 32 + t * 16 + l16;
            int kc2 = (key < NTOT) ? key : (NTOT - 1);
            const half_t* kp = kH + base + (long long)kc2 * D_MODEL;
            #pragma unroll
            for (int kc = 0; kc < 2; ++kc) {
                const v8h* bp = (const v8h*)(kp + kc * 32 + hi * 16);
                v16h bf = SHUF16(bp[0], bp[1]);
                S[t] = __builtin_amdgcn_wmma_f32_16x16x32_f16(
                    false, qa[kc], false, bf, (short)0, S[t], false, false);
            }
        }
        // mask + online softmax per row (rows g + 8*hi, cols across 16 lanes)
        int k0i = kb * 32 + l16;
        int k1i = k0i + 16;
        int k0c = (k0i < NTOT) ? k0i : (NTOT - 1);
        int k1c = (k1i < NTOT) ? k1i : (NTOT - 1);
        float p0g[8], p1g[8];
        #pragma unroll
        for (int g = 0; g < 8; ++g) {
            int q  = q0 + g + hi * 8;
            int qc = (q < NTOT) ? q : (NTOT - 1);
            const unsigned char* brow = blocked + (long long)qc * NTOT;
            bool bad0 = (q >= NTOT) || (k0i >= NTOT) || brow[k0c];
            bool bad1 = (q >= NTOT) || (k1i >= NTOT) || brow[k1c];
            float s0 = bad0 ? -1e9f : S[0][g] * scale;
            float s1 = bad1 ? -1e9f : S[1][g] * scale;
            float mx = fmaxf(s0, s1);
            for (int msk = 1; msk < 16; msk <<= 1) mx = fmaxf(mx, __shfl_xor(mx, msk, 32));
            float mnew = fmaxf(m_[g], mx);
            float sc   = __expf(m_[g] - mnew);
            float p0   = __expf(s0 - mnew);
            float p1   = __expf(s1 - mnew);
            float ls   = p0 + p1;
            for (int msk = 1; msk < 16; msk <<= 1) ls += __shfl_xor(ls, msk, 32);
            l_[g] = l_[g] * sc + ls;
            m_[g] = mnew;
            p0g[g] = p0; p1g[g] = p1;
            #pragma unroll
            for (int nt = 0; nt < 4; ++nt) O[nt][g] *= sc;
        }
        // P (C-layout) -> LDS -> A-layout fragment
        __syncthreads();
        #pragma unroll
        for (int g = 0; g < 8; ++g) {
            int row = g + hi * 8;
            Plds[row * 32 + l16]      = (half_t)p0g[g];
            Plds[row * 32 + 16 + l16] = (half_t)p1g[g];
        }
        __syncthreads();
        v16h pa;
        #pragma unroll
        for (int e = 0; e < 16; ++e) {
            int kk = e + (hi ? 16 : 0);
            pa[e] = Plds[l16 * 32 + kk];
        }
        // O += P * V   (V transposed, ld = NTOT_PAD, pad rows are zero)
        #pragma unroll
        for (int nt = 0; nt < 4; ++nt) {
            const half_t* vtp = vT + ((long long)b * D_MODEL + h * D_K + nt * 16 + l16) * NTOT_PAD
                                   + kb * 32 + hi * 16;
            const v8h* vp = (const v8h*)vtp;
            v16h vb = SHUF16(vp[0], vp[1]);
            O[nt] = __builtin_amdgcn_wmma_f32_16x16x32_f16(
                false, pa, false, vb, (short)0, O[nt], false, false);
        }
    }
    // normalize + store
    #pragma unroll
    for (int nt = 0; nt < 4; ++nt)
        #pragma unroll
        for (int g = 0; g < 8; ++g) {
            int q = q0 + g + hi * 8;
            if (q < NTOT) {
                float v = O[nt][g] / l_[g];
                oH[base + (long long)q * D_MODEL + nt * 16 + l16] = (half_t)v;
            }
        }
}

// ---------------------------------------------------------------------------
// Zero the transposed-V padding rows (tokens 1361..1375), so PV reads are safe
// ---------------------------------------------------------------------------
__global__ void zero_vT_pad(half_t* __restrict__ vT) {
    int idx = blockIdx.x * 256 + threadIdx.x;
    const int npad = NTOT_PAD - NTOT;                 // 15
    const int total = BATCH * D_MODEL * npad;
    if (idx >= total) return;
    int p = idx % npad;
    int d = (idx / npad) % D_MODEL;
    int b = idx / (npad * D_MODEL);
    vT[((long long)b * D_MODEL + d) * NTOT_PAD + NTOT + p] = (half_t)0.0f;
}

// ---------------------------------------------------------------------------
// LayerNorm over 512 cols, one wave per row, optional residual, dual output
// ---------------------------------------------------------------------------
__global__ __launch_bounds__(128)
void layernorm_kernel(const float* __restrict__ x, const float* __restrict__ res,
                      const float* __restrict__ s, const float* __restrict__ bb,
                      float* __restrict__ outF, half_t* __restrict__ outH, int rows) {
    int wave = threadIdx.x >> 5;
    int lane = threadIdx.x & 31;
    int row  = blockIdx.x * 4 + wave;
    if (row >= rows) return;
    const float* xr = x + (long long)row * D_MODEL;
    const float* rr = res ? res + (long long)row * D_MODEL : nullptr;
    float v[16], sum = 0.0f, sq = 0.0f;
    #pragma unroll
    for (int i = 0; i < 16; ++i) {
        float t = xr[lane + 32 * i];
        if (rr) t += rr[lane + 32 * i];
        v[i] = t; sum += t; sq += t * t;
    }
    for (int m = 1; m < 32; m <<= 1) { sum += __shfl_xor(sum, m, 32); sq += __shfl_xor(sq, m, 32); }
    float mean = sum / (float)D_MODEL;
    float var  = sq / (float)D_MODEL - mean * mean;
    float inv  = rsqrtf(var + 1e-5f);
    #pragma unroll
    for (int i = 0; i < 16; ++i) {
        int c = lane + 32 * i;
        float y = (v[i] - mean) * inv * s[c] + bb[c];
        if (outF) outF[(long long)row * D_MODEL + c] = y;
        if (outH) outH[(long long)row * D_MODEL + c] = (half_t)y;
    }
}

// ---------------------------------------------------------------------------
// Gather final prediction input rows {1024,1280,1344,1360} per batch -> f16
// ---------------------------------------------------------------------------
__global__ void gather_last_kernel(const float* __restrict__ seq, half_t* __restrict__ lastH) {
    int idx = blockIdx.x * 256 + threadIdx.x;     // 4 * 4 * 512
    if (idx >= BATCH * 4 * D_MODEL) return;
    int d = idx % D_MODEL;
    int j = (idx / D_MODEL) & 3;
    int b = idx / (4 * D_MODEL);
    const int rows[4] = {1024, 1280, 1344, 1360};
    lastH[idx] = (half_t)seq[((long long)b * NTOT + rows[j]) * D_MODEL + d];
}

// ---------------------------------------------------------------------------
// Host orchestration
// ---------------------------------------------------------------------------
static inline dim3 g1(long long n) { return dim3((unsigned)((n + 255) / 256)); }

extern "C" void kernel_launch(void* const* d_in, const int* in_sizes, int n_in,
                              void* d_out, int out_size, void* d_ws, size_t ws_size,
                              hipStream_t stream) {
    (void)in_sizes; (void)n_in; (void)out_size;
    const float* x_enc   = (const float*)d_in[0];
    const float* xme     = (const float*)d_in[1];
    const float* xmd     = (const float*)d_in[3];
    const float* token_w = (const float*)d_in[4];
    const float* token_b = (const float*)d_in[5];
    const float* temp_w  = (const float*)d_in[6];
    const float* temp_b  = (const float*)d_in[7];
    const float* down_w  = (const float*)d_in[8];
    const float* down_b  = (const float*)d_in[9];
    const float* conv_w  = (const float*)d_in[10];
    const float* conv_b  = (const float*)d_in[11];
    const float* up_w    = (const float*)d_in[12];
    const float* up_b    = (const float*)d_in[13];
    const float* cln_s   = (const float*)d_in[14];
    const float* cln_b   = (const float*)d_in[15];
    const float* Wq      = (const float*)d_in[16];
    const float* Wk      = (const float*)d_in[17];
    const float* Wv      = (const float*)d_in[18];
    const float* Wo      = (const float*)d_in[19];
    const float* bo      = (const float*)d_in[20];
    const float* ln1_s   = (const float*)d_in[21];
    const float* ln1_b   = (const float*)d_in[22];
    const float* W1      = (const float*)d_in[23];
    const float* b1      = (const float*)d_in[24];
    const float* W2      = (const float*)d_in[25];
    const float* b2      = (const float*)d_in[26];
    const float* ln2_s   = (const float*)d_in[27];
    const float* ln2_b   = (const float*)d_in[28];
    const float* pred_w  = (const float*)d_in[29];
    float* out = (float*)d_out;

    // bump allocator over workspace
    char* ws = (char*)d_ws;
    size_t off = 0;
    auto alloc = [&](size_t bytes) -> char* {
        char* p = ws + off;
        off += (bytes + 255) & ~(size_t)255;
        return p;
    };
    const long long SEQ_ELEMS = (long long)BATCH * NTOT * D_MODEL;   // 2,787,328
    const int PRED_PAD = (PRED_OUT + 63) & ~63;                      // 704
    unsigned char* blocked = (unsigned char*)alloc((size_t)NTOT * NTOT);
    float*  seqF = (float*) alloc(SEQ_ELEMS * 4);
    half_t* seqH = (half_t*)alloc(SEQ_ELEMS * 2);
    half_t* wqH  = (half_t*)alloc((size_t)E_LAYERS * D_MODEL * D_MODEL * 2);
    half_t* wkH  = (half_t*)alloc((size_t)E_LAYERS * D_MODEL * D_MODEL * 2);
    half_t* wvH  = (half_t*)alloc((size_t)E_LAYERS * D_MODEL * D_MODEL * 2);
    half_t* woH  = (half_t*)alloc((size_t)E_LAYERS * D_MODEL * D_MODEL * 2);
    half_t* w1H  = (half_t*)alloc((size_t)E_LAYERS * D_MODEL * D_FF * 2);
    half_t* w2H  = (half_t*)alloc((size_t)E_LAYERS * D_FF * D_MODEL * 2);
    half_t* downH= (half_t*)alloc((size_t)D_MODEL * D_BN * 2);
    half_t* upH  = (half_t*)alloc((size_t)D_BN * D_MODEL * 2);
    half_t* predH= (half_t*)alloc((size_t)4 * D_MODEL * PRED_PAD * 2);
    half_t* lastH= (half_t*)alloc((size_t)BATCH * 4 * D_MODEL * 2);
    half_t* qHb  = (half_t*)alloc(SEQ_ELEMS * 2);
    half_t* kHb  = (half_t*)alloc(SEQ_ELEMS * 2);
    half_t* vT   = (half_t*)alloc((size_t)BATCH * D_MODEL * NTOT_PAD * 2);
    half_t* oHb  = (half_t*)alloc(SEQ_ELEMS * 2);
    float*  tmp1 = (float*) alloc(SEQ_ELEMS * 4);
    half_t* ffnH = (half_t*)alloc((size_t)BATCH * NTOT * D_FF * 2);
    if (ws_size < off) return;   // not enough scratch; deterministic no-op

    // phase-1 aliases (lifetimes disjoint from q/k/vT usage)
    half_t* embH  = qHb;                                   // 4*1025*512 f16
    float*  c0    = (float*)kHb;                           // 4*1025*128 f32
    float*  cpyr  = (float*)vT;                            // 4*336*128 f32
    half_t* cpyrH = (half_t*)((char*)vT + (1 << 20));      // after cpyr

    // 0. mask + weight packs (into WMMA B-fragment order)
    build_mask_kernel<<<g1((long long)NTOT * NTOT), 256, 0, stream>>>(blocked);
    for (int i = 0; i < E_LAYERS; ++i) {
        const size_t wOff  = (size_t)i * D_MODEL * D_MODEL;
        const size_t w1Off = (size_t)i * D_MODEL * D_FF;
        const size_t w2Off = (size_t)i * D_FF * D_MODEL;
        pack_b_kernel<<<g1((long long)D_MODEL*D_MODEL), 256, 0, stream>>>(Wq + wOff,  wqH + wOff, D_MODEL, D_MODEL, D_MODEL);
        pack_b_kernel<<<g1((long long)D_MODEL*D_MODEL), 256, 0, stream>>>(Wk + wOff,  wkH + wOff, D_MODEL, D_MODEL, D_MODEL);
        pack_b_kernel<<<g1((long long)D_MODEL*D_MODEL), 256, 0, stream>>>(Wv + wOff,  wvH + wOff, D_MODEL, D_MODEL, D_MODEL);
        pack_b_kernel<<<g1((long long)D_MODEL*D_MODEL), 256, 0, stream>>>(Wo + wOff,  woH + wOff, D_MODEL, D_MODEL, D_MODEL);
        pack_b_kernel<<<g1((long long)D_MODEL*D_FF),    256, 0, stream>>>(W1 + w1Off, w1H + w1Off, D_MODEL, D_FF, D_FF);
        pack_b_kernel<<<g1((long long)D_FF*D_MODEL),    256, 0, stream>>>(W2 + w2Off, w2H + w2Off, D_FF, D_MODEL, D_MODEL);
    }
    pack_b_kernel<<<g1((long long)D_MODEL*D_BN), 256, 0, stream>>>(down_w, downH, D_MODEL, D_BN, D_BN);
    pack_b_kernel<<<g1((long long)D_BN*D_MODEL), 256, 0, stream>>>(up_w,   upH,   D_BN, D_MODEL, D_MODEL);
    pack_b_kernel<<<g1((long long)4*D_MODEL*PRED_PAD), 256, 0, stream>>>(pred_w, predH, 4*D_MODEL, PRED_OUT, PRED_PAD);

    // 1. embedding into seq[:, :1025, :]
    embed_kernel<<<g1((long long)BATCH*LSEQ*D_MODEL), 256, 0, stream>>>(
        x_enc, xme, xmd, token_w, token_b, temp_w, temp_b, seqF);

    // 2. CSCM pyramid
    const long long sbSeq = (long long)NTOT * D_MODEL;
    cast_f32_f16_strided<<<g1((long long)BATCH*LSEQ*D_MODEL), 256, 0, stream>>>(
        seqF, embH, BATCH * LSEQ, D_MODEL, LSEQ, sbSeq);
    {   // down: (B*1025, 512) x (512, 128)
        int M = BATCH * LSEQ;
        gemm_f16_wmma<<<dim3((D_BN+63)/64, (M+63)/64), 128, 0, stream>>>(
            embH, downH, down_b, c0, nullptr, nullptr, 0, 0, M, D_BN, D_MODEL, M, 0, M, 0, 0);
    }
    conv_elu_kernel<<<g1((long long)BATCH*256*D_BN), 256, 0, stream>>>(
        c0, conv_w + 0 * D_BN * D_BN * 4, conv_b + 0, cpyr, 256, (long long)LSEQ * D_BN, 336LL * D_BN, 0);
    conv_elu_kernel<<<g1((long long)BATCH*64*D_BN), 256, 0, stream>>>(
        cpyr, conv_w + 1 * D_BN * D_BN * 4, conv_b + D_BN, cpyr, 64, 336LL * D_BN, 336LL * D_BN, 256LL * D_BN);
    conv_elu_kernel<<<g1((long long)BATCH*16*D_BN), 256, 0, stream>>>(
        cpyr + 256LL * D_BN, conv_w + 2 * D_BN * D_BN * 4, conv_b + 2 * D_BN, cpyr, 16, 336LL * D_BN, 336LL * D_BN, 320LL * D_BN);
    cast_f32_f16<<<g1((long long)BATCH*336*D_BN), 256, 0, stream>>>(cpyr, cpyrH, (long long)BATCH*336*D_BN);
    {   // up: (B*336, 128) x (128, 512) -> scattered into seq[:, 1025:, :]
        int M = BATCH * 336;
        gemm_f16_wmma<<<dim3((D_MODEL+63)/64, (M+63)/64), 128, 0, stream>>>(
            cpyrH, upH, up_b, seqF + (long long)LSEQ * D_MODEL, nullptr, nullptr, 0, 0,
            M, D_MODEL, D_BN, M, 0, 336, sbSeq, 0);
    }
    // 3. cscm layernorm (in place) -> seqF + seqH
    {
        int rows = BATCH * NTOT;
        layernorm_kernel<<<(rows + 3) / 4, 128, 0, stream>>>(seqF, nullptr, cln_s, cln_b, seqF, seqH, rows);
    }

    // 4. encoder layers
    const int M  = BATCH * NTOT;                 // 5444
    const dim3 gQKV((D_MODEL + 63) / 64, (M + 63) / 64);
    const dim3 gFF1((D_FF + 63) / 64,    (M + 63) / 64);
    for (int i = 0; i < E_LAYERS; ++i) {
        const half_t* wq = wqH + (size_t)i * D_MODEL * D_MODEL;
        const half_t* wk = wkH + (size_t)i * D_MODEL * D_MODEL;
        const half_t* wv = wvH + (size_t)i * D_MODEL * D_MODEL;
        const half_t* wo = woH + (size_t)i * D_MODEL * D_MODEL;
        gemm_f16_wmma<<<gQKV, 128, 0, stream>>>(seqH, wq, nullptr, nullptr, qHb, nullptr, 0, 0,
                                                M, D_MODEL, D_MODEL, M, 0, M, 0, 0);
        gemm_f16_wmma<<<gQKV, 128, 0, stream>>>(seqH, wk, nullptr, nullptr, kHb, nullptr, 0, 0,
                                                M, D_MODEL, D_MODEL, M, 0, M, 0, 0);
        zero_vT_pad<<<g1((long long)BATCH * D_MODEL * (NTOT_PAD - NTOT)), 256, 0, stream>>>(vT);
        gemm_f16_wmma<<<gQKV, 128, 0, stream>>>(seqH, wv, nullptr, nullptr, nullptr,
                                                vT, NTOT_PAD, (long long)D_MODEL * NTOT_PAD,
                                                M, D_MODEL, D_MODEL, M, 0, NTOT, 0, 0);
        attn_wmma<<<dim3((NTOT + 15) / 16, BATCH * N_HEADS), 32, 0, stream>>>(qHb, kHb, vT, blocked, oHb);
        gemm_f16_wmma<<<gQKV, 128, 0, stream>>>(oHb, wo, bo + i * D_MODEL, tmp1, nullptr, nullptr, 0, 0,
                                                M, D_MODEL, D_MODEL, M, 0, M, 0, 0);
        layernorm_kernel<<<(M + 3) / 4, 128, 0, stream>>>(tmp1, seqF, ln1_s + i * D_MODEL, ln1_b + i * D_MODEL, seqF, seqH, M);
        gemm_f16_wmma<<<gFF1, 128, 0, stream>>>(seqH, w1H + (size_t)i * D_MODEL * D_FF, b1 + i * D_FF,
                                                nullptr, ffnH, nullptr, 0, 0,
                                                M, D_FF, D_MODEL, M, 0, M, 0, /*GELU*/1);
        gemm_f16_wmma<<<gQKV, 128, 0, stream>>>(ffnH, w2H + (size_t)i * D_FF * D_MODEL, b2 + i * D_MODEL,
                                                tmp1, nullptr, nullptr, 0, 0,
                                                M, D_MODEL, D_FF, M, 0, M, 0, 0);
        layernorm_kernel<<<(M + 3) / 4, 128, 0, stream>>>(tmp1, seqF, ln2_s + i * D_MODEL, ln2_b + i * D_MODEL, seqF, seqH, M);
    }

    // 5. prediction head: gather 4 pyramid rows per batch -> (4, 2048) x (2048, 672)
    gather_last_kernel<<<g1(BATCH * 4 * D_MODEL), 256, 0, stream>>>(seqF, lastH);
    gemm_f16_wmma<<<dim3((PRED_OUT + 63) / 64, 1), 128, 0, stream>>>(
        lastH, predH, nullptr, out, nullptr, nullptr, 0, 0,
        BATCH, PRED_OUT, 4 * D_MODEL, BATCH, 0, BATCH, 0, 0);
}
